// Block_87514253623489
// MI455X (gfx1250) — matrix operand order repaired and checked by reading, not compiled, not run on hardware
//
#include <hip/hip_runtime.h>
#include <hip/hip_bf16.h>
#include <math.h>

// ---------------------------------------------------------------------------
// Types / helpers
// ---------------------------------------------------------------------------
typedef unsigned int   u32;
typedef unsigned short u16;
typedef __attribute__((ext_vector_type(16))) __bf16 v16bf;
typedef __attribute__((ext_vector_type(8)))  float  v8f;
typedef int v4i __attribute__((vector_size(16)));   // matches builtin param type

struct Frag8 { u32 d[8]; };   // 32 bytes == 16 bf16 (one WMMA A/B operand, wave32)

__device__ __forceinline__ u16 f2bf(float f) {
    u32 u = __builtin_bit_cast(u32, f);
    u += 0x7fffu + ((u >> 16) & 1u);   // round-to-nearest-even
    return (u16)(u >> 16);
}

// Build one 16-bf16 fragment from two contiguous 16B LDS reads.
__device__ __forceinline__ v16bf ld_frag2(const u16* p0, const u16* p1) {
    Frag8 f;
    uint4 a = *reinterpret_cast<const uint4*>(p0);
    uint4 b = *reinterpret_cast<const uint4*>(p1);
    f.d[0] = a.x; f.d[1] = a.y; f.d[2] = a.z; f.d[3] = a.w;
    f.d[4] = b.x; f.d[5] = b.y; f.d[6] = b.z; f.d[7] = b.w;
    return __builtin_bit_cast(v16bf, f);
}

__device__ __forceinline__ v8f wmma_bf16(v16bf a, v16bf b, v8f c) {
    // D = A(16x32 bf16) * B(32x16 bf16) + C(16x16 f32)
    return __builtin_amdgcn_wmma_f32_16x16x32_bf16(
        /*neg_a=*/false, a, /*neg_b=*/false, b,
        /*c_mod=*/(short)0, c, /*reuse_a=*/false, /*reuse_b=*/false);
}

__device__ __forceinline__ v8f zero8() {
    v8f v;
#pragma unroll
    for (int i = 0; i < 8; ++i) v[i] = 0.f;
    return v;
}

// ---------------------------------------------------------------------------
// CDNA5 async Global -> LDS copy (ASYNCcnt-tracked DMA), guarded so the file
// still compiles on toolchains that lack the builtin (sync uint4 fallback).
// Builtin signature (from hipcc diagnostic): (v4i addrspace(1)*, v4i addrspace(3)*, imm, imm)
// ---------------------------------------------------------------------------
#if __has_builtin(__builtin_amdgcn_global_load_async_to_lds_b128)
#define HAVE_ASYNC_LDS 1
#endif

__device__ __forceinline__ void async_copy16(const u16* __restrict__ gsrc,
                                             u16* __restrict__ ldst) {
#ifdef HAVE_ASYNC_LDS
    __builtin_amdgcn_global_load_async_to_lds_b128(
        (__attribute__((address_space(1))) v4i*)gsrc,
        (__attribute__((address_space(3))) v4i*)ldst,
        /*imm offset=*/0, /*cpol=*/0);
#else
    *reinterpret_cast<uint4*>(ldst) = *reinterpret_cast<const uint4*>(gsrc);
#endif
}

__device__ __forceinline__ void async_wait() {
#ifdef HAVE_ASYNC_LDS
#if __has_builtin(__builtin_amdgcn_s_wait_asynccnt)
    __builtin_amdgcn_s_wait_asynccnt(0);
#endif
#endif
}

// ---------------------------------------------------------------------------
// fp32 -> bf16 conversion (weights, /4 vectors)
// ---------------------------------------------------------------------------
__global__ void __launch_bounds__(256)
cvt_f32_bf16(const float* __restrict__ in, u16* __restrict__ out, int n) {
    int i = (blockIdx.x * 256 + threadIdx.x) * 4;
    if (i + 3 < n) {
        float4 v = *reinterpret_cast<const float4*>(in + i);
        u32 p0 = (u32)f2bf(v.x) | ((u32)f2bf(v.y) << 16);
        u32 p1 = (u32)f2bf(v.z) | ((u32)f2bf(v.w) << 16);
        *reinterpret_cast<uint2*>(out + i) = make_uint2(p0, p1);
    }
}

// ---------------------------------------------------------------------------
// LayerNorm over D=1024, one row per block (256 threads, 4 floats/thread),
// emits bf16 for WMMA consumption.
// ---------------------------------------------------------------------------
__global__ void __launch_bounds__(256)
layernorm_bf16(const float* __restrict__ x, const float* __restrict__ w,
               const float* __restrict__ b, u16* __restrict__ out) {
    __shared__ float red[16];
    const int row = blockIdx.x;
    const int tid = threadIdx.x;
    const float4 v = reinterpret_cast<const float4*>(x + (size_t)row * 1024)[tid];
    float s  = v.x + v.y + v.z + v.w;
    float s2 = v.x * v.x + v.y * v.y + v.z * v.z + v.w * v.w;
#pragma unroll
    for (int off = 16; off; off >>= 1) {
        s  += __shfl_xor(s,  off, 32);
        s2 += __shfl_xor(s2, off, 32);
    }
    const int wave = tid >> 5, lane = tid & 31;
    if (lane == 0) { red[wave] = s; red[8 + wave] = s2; }
    __syncthreads();
    if (tid == 0) {
        float ts = 0.f, ts2 = 0.f;
#pragma unroll
        for (int i = 0; i < 8; ++i) { ts += red[i]; ts2 += red[8 + i]; }
        red[0] = ts; red[8] = ts2;
    }
    __syncthreads();
    const float mu   = red[0] * (1.f / 1024.f);
    const float var  = red[8] * (1.f / 1024.f) - mu * mu;
    const float rstd = rsqrtf(var + 1e-5f);
    const float4 wv = reinterpret_cast<const float4*>(w)[tid];
    const float4 bv = reinterpret_cast<const float4*>(b)[tid];
    u32 p0 = (u32)f2bf((v.x - mu) * rstd * wv.x + bv.x) |
             ((u32)f2bf((v.y - mu) * rstd * wv.y + bv.y) << 16);
    u32 p1 = (u32)f2bf((v.z - mu) * rstd * wv.z + bv.z) |
             ((u32)f2bf((v.w - mu) * rstd * wv.w + bv.w) << 16);
    *reinterpret_cast<uint2*>(out + (size_t)row * 1024 + tid * 4) = make_uint2(p0, p1);
}

// ---------------------------------------------------------------------------
// Tiled bf16 WMMA GEMM: C[M,N] = A[M,K] * B[K,N] (+bias) with fused epilogues.
//   epi 0: fp32 store, optional +residual     (proj, fc2)
//   epi 1: exact GELU -> bf16 store           (fc1)
//   epi 2: QKV split -> bf16 [B,H,T,Dh] q/k/v (qkv)
// Block tile 128x128, BK=64 (two WMMA k-steps per stage),
// 8 waves: 4 (M) x 2 (N), wave tile 32x64 -> 16 WMMAs per barrier pair.
// ---------------------------------------------------------------------------
#define BM 128
#define BN 128
#define BK 64

__global__ void __launch_bounds__(256)
gemm_bf16(const u16* __restrict__ A, const u16* __restrict__ B,
          const float* __restrict__ bias, int M, int N, int K, int epi,
          float* __restrict__ outF, u16* __restrict__ outH,
          const float* __restrict__ residual,
          u16* __restrict__ q_out, u16* __restrict__ k_out, u16* __restrict__ v_out) {
    __shared__ u16 As[BM * BK];   // [m][k]   16KB
    __shared__ u16 Bs[BN * BK];   // [n][k]   16KB (transposed at staging time)

    const int tid  = threadIdx.x;
    const int lane = tid & 31;
    const int wave = tid >> 5;
    const int wm   = wave & 3;       // M direction, 32 rows each
    const int wn   = wave >> 2;      // N direction, 64 cols each
    const int half = lane >> 4;
    const int lr   = lane & 15;
    const int m0   = blockIdx.y * BM;
    const int n0   = blockIdx.x * BN;

    v8f acc[2][4];
#pragma unroll
    for (int mi = 0; mi < 2; ++mi)
#pragma unroll
        for (int ni = 0; ni < 4; ++ni) acc[mi][ni] = zero8();

    for (int kk = 0; kk < K; kk += BK) {
        // --- stage A tile 128x64 bf16 via async Global->LDS DMA ---
#pragma unroll
        for (int j = 0; j < 4; ++j) {
            int id = tid + j * 256;
            int r = id >> 3, c = (id & 7) * 8;
            async_copy16(A + (size_t)(m0 + r) * K + kk + c, &As[r * BK + c]);
        }
        // --- stage B tile 64x128 transposed to [n][k] (manual: needs transpose) ---
#pragma unroll
        for (int j = 0; j < 4; ++j) {
            int id = tid + j * 256;
            int kr = id >> 4, c = (id & 15) * 8;
            uint4 vv = *reinterpret_cast<const uint4*>(B + (size_t)(kk + kr) * N + n0 + c);
            const u16* e = reinterpret_cast<const u16*>(&vv);
#pragma unroll
            for (int x = 0; x < 8; ++x) Bs[(c + x) * BK + kr] = e[x];
        }
        // --- prefetch next stage into cache ---
        if (kk + BK < K) {
            int r = tid >> 3, c = (tid & 7) * 8;
            __builtin_prefetch(A + (size_t)(m0 + r) * K + kk + BK + c, 0, 0);
            int kr = tid >> 4, c2 = (tid & 15) * 8;
            __builtin_prefetch(B + (size_t)(kk + BK + kr) * N + n0 + c2, 0, 0);
        }
        async_wait();
        __syncthreads();

#pragma unroll
        for (int ks = 0; ks < BK; ks += 32) {
            v16bf af[2], bfr[4];
#pragma unroll
            for (int mi = 0; mi < 2; ++mi) {
                int row = wm * 32 + mi * 16 + lr;
                // A 16x32 bf16 layout: half0 lanes: K 0..7 & 16..23; half1: K 8..15 & 24..31
                af[mi] = ld_frag2(&As[row * BK + ks + half * 8],
                                  &As[row * BK + ks + 16 + half * 8]);
            }
#pragma unroll
            for (int ni = 0; ni < 4; ++ni) {
                int col = wn * 64 + ni * 16 + lr;
                // B 32x16 bf16 layout: half0 lanes: K 0..7 & 16..23; half1: K 8..15 & 24..31
                bfr[ni] = ld_frag2(&Bs[col * BK + ks + half * 8],
                                   &Bs[col * BK + ks + 16 + half * 8]);
            }
#pragma unroll
            for (int mi = 0; mi < 2; ++mi)
#pragma unroll
                for (int ni = 0; ni < 4; ++ni)
                    acc[mi][ni] = wmma_bf16(af[mi], bfr[ni], acc[mi][ni]);
        }
        __syncthreads();
    }

    // --- epilogue: C layout: VGPR i -> M = i + 8*half, N = lane%16 ---
#pragma unroll
    for (int mi = 0; mi < 2; ++mi) {
#pragma unroll
        for (int ni = 0; ni < 4; ++ni) {
            const int col = n0 + wn * 64 + ni * 16 + lr;
            const float bb = bias ? bias[col] : 0.f;
#pragma unroll
            for (int i = 0; i < 8; ++i) {
                const int row = m0 + wm * 32 + mi * 16 + i + half * 8;
                float val = acc[mi][ni][i] + bb;
                if (epi == 0) {
                    if (residual) val += residual[(size_t)row * N + col];
                    outF[(size_t)row * N + col] = val;
                } else if (epi == 1) {
                    float g = 0.5f * val * (1.f + erff(val * 0.70710678118654752f));
                    outH[(size_t)row * N + col] = f2bf(g);
                } else {
                    const int which = col >> 10;          // 0=q 1=k 2=v
                    const int h     = (col & 1023) >> 6;  // head
                    const int d     = col & 63;
                    const int bidx  = row >> 11;          // batch (T=2048)
                    const int t     = row & 2047;
                    u16* dst = (which == 0) ? q_out : (which == 1) ? k_out : v_out;
                    dst[(((size_t)(bidx * 16 + h)) * 2048 + t) * 64 + d] = f2bf(val);
                }
            }
        }
    }
}

// ---------------------------------------------------------------------------
// Flash attention (causal), bf16 WMMA. One block per (qb of 128 rows, b*h).
// 8 waves, each owns 16 q rows. Key blocks of 64. Dh = 64, scale = 1/8.
// q/k/v layout: [B*H, T, Dh] bf16. Output: [B*T, H*Dh] bf16.
// ---------------------------------------------------------------------------
__global__ void __launch_bounds__(256)
flash_attn(const u16* __restrict__ Q, const u16* __restrict__ Kg,
           const u16* __restrict__ Vg, u16* __restrict__ Out) {
    __shared__ u16 Qs[128 * 64];   // [q][d]     16KB
    __shared__ u16 Ks[64 * 64];    // [key][d]    8KB
    __shared__ u16 Vs[64 * 64];    // [d][key]    8KB (transposed for B-frags)
    __shared__ u16 Ps[128 * 64];   // [q][key]   16KB

    const int tid  = threadIdx.x;
    const int lane = tid & 31;
    const int wave = tid >> 5;
    const int half = lane >> 4;
    const int lr   = lane & 15;
    const int qb   = blockIdx.x;   // 0..15
    const int bh   = blockIdx.y;   // 0..31
    const size_t base = (size_t)bh * 2048 * 64;

    // stage Q once (async DMA)
#pragma unroll
    for (int j = 0; j < 2; ++j) {
        int id = tid + j * 256;
        int r = id >> 3, c = (id & 7) * 8;
        async_copy16(Q + base + (size_t)(qb * 128 + r) * 64 + c, &Qs[r * 64 + c]);
    }
    async_wait();
    __syncthreads();

    const int ql = wave * 16;      // this wave's local q-row base
    v16bf aq[2];
    {
        int row = ql + lr;
        aq[0] = ld_frag2(&Qs[row * 64 + half * 8],      &Qs[row * 64 + 16 + half * 8]);
        aq[1] = ld_frag2(&Qs[row * 64 + 32 + half * 8], &Qs[row * 64 + 48 + half * 8]);
    }

    v8f o[4];
#pragma unroll
    for (int ni = 0; ni < 4; ++ni) o[ni] = zero8();
    float m_i[8], l_i[8];
#pragma unroll
    for (int i = 0; i < 8; ++i) { m_i[i] = -1e30f; l_i[i] = 0.f; }

    const int kbmax = qb * 2 + 1;  // causal: keys up to qb*128+127
    for (int kb = 0; kb <= kbmax; ++kb) {
        // stage K [key][d] (async) and V transposed [d][key] (manual)
#pragma unroll
        for (int j = 0; j < 2; ++j) {
            int id = tid + j * 256;
            int r = id >> 3, c = (id & 7) * 8;
            async_copy16(Kg + base + (size_t)(kb * 64 + r) * 64 + c, &Ks[r * 64 + c]);
            uint4 vv = *reinterpret_cast<const uint4*>(Vg + base + (size_t)(kb * 64 + r) * 64 + c);
            const u16* e = reinterpret_cast<const u16*>(&vv);
#pragma unroll
            for (int x = 0; x < 8; ++x) Vs[(c + x) * 64 + r] = e[x];
        }
        async_wait();
        __syncthreads();

        // S = Q * K^T (scale 1/8), causal mask, online softmax
        float s_all[4][8];
        float rmax[8];
#pragma unroll
        for (int i = 0; i < 8; ++i) rmax[i] = -1e30f;
#pragma unroll
        for (int ni = 0; ni < 4; ++ni) {
            int key = ni * 16 + lr;
            v16bf b0 = ld_frag2(&Ks[key * 64 + half * 8],      &Ks[key * 64 + 16 + half * 8]);
            v16bf b1 = ld_frag2(&Ks[key * 64 + 32 + half * 8], &Ks[key * 64 + 48 + half * 8]);
            v8f s = zero8();
            s = wmma_bf16(aq[0], b0, s);
            s = wmma_bf16(aq[1], b1, s);
            int kidx = kb * 64 + ni * 16 + lr;
#pragma unroll
            for (int i = 0; i < 8; ++i) {
                int qidx = qb * 128 + ql + i + half * 8;
                float sv = (kidx <= qidx) ? s[i] * 0.125f : -1e30f;
                s_all[ni][i] = sv;
                rmax[i] = fmaxf(rmax[i], sv);
            }
        }
#pragma unroll
        for (int i = 0; i < 8; ++i) {   // row-max across the 16 N-lanes of each half
            float r = rmax[i];
            r = fmaxf(r, __shfl_xor(r, 1, 32));
            r = fmaxf(r, __shfl_xor(r, 2, 32));
            r = fmaxf(r, __shfl_xor(r, 4, 32));
            r = fmaxf(r, __shfl_xor(r, 8, 32));
            rmax[i] = r;
        }
        float alpha[8], rs[8];
#pragma unroll
        for (int i = 0; i < 8; ++i) {
            float mn = fmaxf(m_i[i], rmax[i]);
            alpha[i] = __expf(m_i[i] - mn);
            m_i[i] = mn;
            rs[i] = 0.f;
        }
#pragma unroll
        for (int ni = 0; ni < 4; ++ni)
#pragma unroll
            for (int i = 0; i < 8; ++i) {
                float p = __expf(s_all[ni][i] - m_i[i]);
                s_all[ni][i] = p;
                rs[i] += p;
            }
#pragma unroll
        for (int i = 0; i < 8; ++i) {
            float r = rs[i];
            r += __shfl_xor(r, 1, 32);
            r += __shfl_xor(r, 2, 32);
            r += __shfl_xor(r, 4, 32);
            r += __shfl_xor(r, 8, 32);
            l_i[i] = l_i[i] * alpha[i] + r;
        }
#pragma unroll
        for (int ni = 0; ni < 4; ++ni)
#pragma unroll
            for (int i = 0; i < 8; ++i) o[ni][i] *= alpha[i];

        // spill P (bf16) to LDS to re-enter WMMA A-layout
#pragma unroll
        for (int ni = 0; ni < 4; ++ni)
#pragma unroll
            for (int i = 0; i < 8; ++i)
                Ps[(ql + i + half * 8) * 64 + ni * 16 + lr] = f2bf(s_all[ni][i]);
        __syncthreads();

        // O += P * V
        {
            int row = ql + lr;
            v16bf ap0 = ld_frag2(&Ps[row * 64 + half * 8],      &Ps[row * 64 + 16 + half * 8]);
            v16bf ap1 = ld_frag2(&Ps[row * 64 + 32 + half * 8], &Ps[row * 64 + 48 + half * 8]);
#pragma unroll
            for (int ni = 0; ni < 4; ++ni) {
                int d = ni * 16 + lr;
                v16bf bv0 = ld_frag2(&Vs[d * 64 + half * 8],      &Vs[d * 64 + 16 + half * 8]);
                v16bf bv1 = ld_frag2(&Vs[d * 64 + 32 + half * 8], &Vs[d * 64 + 48 + half * 8]);
                o[ni] = wmma_bf16(ap0, bv0, o[ni]);
                o[ni] = wmma_bf16(ap1, bv1, o[ni]);
            }
        }
        __syncthreads();
    }

    // normalize and store bf16 [B*T, H*Dh]
    const int b = bh >> 4, h = bh & 15;
#pragma unroll
    for (int ni = 0; ni < 4; ++ni)
#pragma unroll
        for (int i = 0; i < 8; ++i) {
            int t   = qb * 128 + ql + i + half * 8;
            int col = h * 64 + ni * 16 + lr;
            Out[((size_t)(b * 2048 + t)) * 1024 + col] = f2bf(o[ni][i] / l_i[i]);
        }
}

// ---------------------------------------------------------------------------
// Orchestration
// ---------------------------------------------------------------------------
extern "C" void kernel_launch(void* const* d_in, const int* in_sizes, int n_in,
                              void* d_out, int out_size, void* d_ws, size_t ws_size,
                              hipStream_t stream) {
    const float* x      = (const float*)d_in[0];
    const float* ln1_w  = (const float*)d_in[1];
    const float* ln1_b  = (const float*)d_in[2];
    const float* attn_w = (const float*)d_in[3];
    const float* attn_b = (const float*)d_in[4];
    const float* proj_w = (const float*)d_in[5];
    const float* proj_b = (const float*)d_in[6];
    const float* ln2_w  = (const float*)d_in[7];
    const float* ln2_b  = (const float*)d_in[8];
    const float* fc_w   = (const float*)d_in[9];
    const float* fc_b   = (const float*)d_in[10];
    const float* fc2_w  = (const float*)d_in[11];
    const float* fc2_b  = (const float*)d_in[12];
    float* out = (float*)d_out;

    const int M = 4096;           // B*T
    const int D = 1024;
    const int F = 4096;           // EXP*D

    char* ws = (char*)d_ws;
    size_t off = 0;
    auto alloc = [&](size_t bytes) -> void* {
        void* p = ws + off;
        off = (off + bytes + 255) & ~(size_t)255;
        return p;
    };
    u16*   wA   = (u16*)alloc((size_t)D * 3 * D * 2);    // attn_w bf16
    u16*   wP   = (u16*)alloc((size_t)D * D * 2);        // proj_w bf16
    u16*   wF1  = (u16*)alloc((size_t)D * F * 2);        // fc_w  bf16
    u16*   wF2  = (u16*)alloc((size_t)F * D * 2);        // fc2_w bf16
    u16*   ln1h = (u16*)alloc((size_t)M * D * 2);
    u16*   qh   = (u16*)alloc((size_t)M * D * 2);        // [B*H, T, 64]
    u16*   kh   = (u16*)alloc((size_t)M * D * 2);
    u16*   vh   = (u16*)alloc((size_t)M * D * 2);
    u16*   ah   = (u16*)alloc((size_t)M * D * 2);        // attn out bf16 [B*T, H*64]
    float* x1   = (float*)alloc((size_t)M * D * 4);      // residual after attn
    u16*   ln2h = (u16*)alloc((size_t)M * D * 2);
    u16*   hh   = (u16*)alloc((size_t)M * F * 2);        // gelu(fc1) bf16

    auto cvt = [&](const float* src, u16* dst, int n) {
        cvt_f32_bf16<<<dim3((n / 4 + 255) / 256), dim3(256), 0, stream>>>(src, dst, n);
    };
    cvt(attn_w, wA,  D * 3 * D);
    cvt(proj_w, wP,  D * D);
    cvt(fc_w,   wF1, D * F);
    cvt(fc2_w,  wF2, F * D);

    // LN1
    layernorm_bf16<<<dim3(M), dim3(256), 0, stream>>>(x, ln1_w, ln1_b, ln1h);

    // QKV: [M,D] x [D,3D] -> q/k/v bf16
    gemm_bf16<<<dim3(3 * D / BN, M / BM), dim3(256), 0, stream>>>(
        ln1h, wA, attn_b, M, 3 * D, D, /*epi=*/2,
        nullptr, nullptr, nullptr, qh, kh, vh);

    // attention
    flash_attn<<<dim3(2048 / 128, 32), dim3(256), 0, stream>>>(qh, kh, vh, ah);

    // proj + residual: x1 = x + ah @ proj_w + proj_b
    gemm_bf16<<<dim3(D / BN, M / BM), dim3(256), 0, stream>>>(
        ah, wP, proj_b, M, D, D, /*epi=*/0,
        x1, nullptr, x, nullptr, nullptr, nullptr);

    // LN2
    layernorm_bf16<<<dim3(M), dim3(256), 0, stream>>>(x1, ln2_w, ln2_b, ln2h);

    // FC1 + exact GELU -> bf16
    gemm_bf16<<<dim3(F / BN, M / BM), dim3(256), 0, stream>>>(
        ln2h, wF1, fc_b, M, F, D, /*epi=*/1,
        nullptr, hh, nullptr, nullptr, nullptr, nullptr);

    // FC2 + residual -> d_out (fp32)
    gemm_bf16<<<dim3(D / BN, M / BM), dim3(256), 0, stream>>>(
        hh, wF2, fc2_b, M, D, F, /*epi=*/0,
        out, nullptr, x1, nullptr, nullptr, nullptr);
}
